// Raindrop_50388556316826
// MI455X (gfx1250) — matrix operand order repaired and checked by reading, not compiled, not run on hardware
//
#include <hip/hip_runtime.h>

// ---------------------------------------------------------------------------
// Types (plain ext-vectors only: trivially-constructible, union-safe, aligned)
// ---------------------------------------------------------------------------
typedef unsigned short u16;
typedef __attribute__((ext_vector_type(16))) __bf16 v16bf;
typedef __attribute__((ext_vector_type(8)))  float  v8f;
typedef __attribute__((ext_vector_type(4)))  unsigned uint4e;
typedef __attribute__((ext_vector_type(4)))  unsigned short us4;

__device__ inline float bf2f(u16 u) { union { float f; unsigned i; } x; x.i = ((unsigned)u) << 16; return x.f; }
__device__ inline u16 f2bf(float f) {
  union { float f; unsigned i; } x; x.f = f;
  unsigned r = x.i + 0x7FFFu + ((x.i >> 16) & 1u);
  return (u16)(r >> 16);
}
__device__ inline int imin(int a, int b) { return a < b ? a : b; }
__device__ inline float wsum(float v) { for (int o = 16; o > 0; o >>= 1) v += __shfl_xor(v, o, 32); return v; }
__device__ inline float wmax(float v) { for (int o = 16; o > 0; o >>= 1) v = fmaxf(v, __shfl_xor(v, o, 32)); return v; }

// ---------------------------------------------------------------------------
// Model dims
// ---------------------------------------------------------------------------
#define SE_   128
#define T_    256
#define B_    128
#define DFEAT 1024
#define DMODEL 528
#define HD    132
#define NH    4
#define DFF   128
#define DFIN  1040

// ---------------------------------------------------------------------------
// Generic WMMA GEMM:  C[z] = scale * (A[z] @ Bw[z]^eff) + bias (+Cold) (+relu)
//   A : bf16 row-major M x K (lda), batch stride sA (elements)
//   Bw: bf16; if b_kxn==0 it is N x K row-major (X @ W^T pattern),
//             if b_kxn==1 it is K x N row-major (plain A@B pattern)
//   out: Cf (f32) or Cb (bf16), row stride ldc, batch stride sC (elements)
// Block = 128 threads (4 waves) -> 64x128 tile; each wave owns 32x64
// (2x4 fragments = 8 x v_wmma_f32_16x16x32_bf16 per 32-wide K step).
// Ragged edges use CLAMPED unconditional loads + select (no exec branching).
// ---------------------------------------------------------------------------
#define TM 64
#define TN 128
#define TK 32

__global__ __launch_bounds__(128)
void k_gemm(const u16* __restrict__ A, int lda, long long sA,
            const u16* __restrict__ Bw, int ldb, long long sB, int b_kxn,
            const float* __restrict__ bias,
            float* __restrict__ Cf, u16* __restrict__ Cb, int ldc, long long sC,
            int M, int N, int K, float scale, int accum, int do_relu)
{
  __shared__ __align__(16) u16 As[TM * TK];
  __shared__ __align__(16) u16 Bs[TN * TK];

  const int tid  = threadIdx.x;
  const int lane = tid & 31;
  const int wid  = tid >> 5;
  const int wy   = wid >> 1;            // 0..1 : 32-row strip
  const int wx   = wid & 1;             // 0..1 : 64-col strip
  const int bm   = blockIdx.y * TM;
  const int bn   = blockIdx.x * TN;
  const int z    = blockIdx.z;

  A  += (long long)z * sA;
  Bw += (long long)z * sB;
  if (Cf) Cf += (long long)z * sC;
  if (Cb) Cb += (long long)z * sC;

  v8f acc[2][4] = {};

  const int ar = tid >> 1;              // 0..63  : A-tile row
  const int ah = (tid & 1) * 16;        // 0/16   : A k-half
  const int br = tid;                   // 0..127 : B-tile row (n)

  for (int k0 = 0; k0 < K; k0 += TK) {
    // ---- A tile (64 x 32): each thread 16 contiguous k
    {
      int row = bm + ar;
      u16* d = &As[ar * TK + ah];
      if (row < M && (k0 + ah + 16) <= K) {
        const u16* s = A + (long long)row * lda + k0 + ah;
        *(us4*)(d)      = *(const us4*)(s);
        *(us4*)(d + 4)  = *(const us4*)(s + 4);
        *(us4*)(d + 8)  = *(const us4*)(s + 8);
        *(us4*)(d + 12) = *(const us4*)(s + 12);
      } else {
        const u16* s = A + (long long)imin(row, M - 1) * lda;  // in-bounds row
        bool rok = row < M;
        #pragma unroll
        for (int j = 0; j < 16; j++) {
          int k = k0 + ah + j;
          u16 v = s[imin(k, K - 1)];                 // unconditional clamped load
          d[j] = (rok && k < K) ? v : (u16)0;        // select, no branch
        }
      }
    }
    // ---- B tile (128 x 32) in LDS layout [n][k]
    {
      int n = bn + br;
      u16* d = &Bs[br * TK];
      if (!b_kxn) {
        if (n < N && (k0 + TK) <= K) {
          const u16* s = Bw + (long long)n * ldb + k0;
          #pragma unroll
          for (int j = 0; j < 8; j++) *(us4*)(d + 4 * j) = *(const us4*)(s + 4 * j);
        } else {
          const u16* s = Bw + (long long)imin(n, N - 1) * ldb;
          bool nok = n < N;
          #pragma unroll
          for (int j = 0; j < TK; j++) {
            int k = k0 + j;
            u16 v = s[imin(k, K - 1)];
            d[j] = (nok && k < K) ? v : (u16)0;
          }
        }
      } else {
        int nc = imin(n, N - 1);
        bool nok = n < N;
        #pragma unroll
        for (int j = 0; j < TK; j++) {
          int k = k0 + j;
          u16 v = Bw[(long long)imin(k, K - 1) * ldb + nc];
          d[j] = (nok && k < K) ? v : (u16)0;
        }
      }
    }
    __syncthreads();

    // ---- fragments per CDNA5 ISA VGPR layouts, two 16B LDS reads each
    const unsigned* As32 = (const unsigned*)As;
    const unsigned* Bs32 = (const unsigned*)Bs;
    const int half = lane >> 4;
    const int l15  = lane & 15;

    union Frag { uint4e u[2]; v16bf v; };
    Frag a[2], b[4];
    #pragma unroll
    for (int i = 0; i < 2; i++) {
      int m = wy * 32 + i * 16 + l15;               // A: lane -> M=l15
      const unsigned* p = As32 + m * 16 + half * 4; // K pairs {0..7|8..15}(+16)
      a[i].u[0] = *(const uint4e*)(p);
      a[i].u[1] = *(const uint4e*)(p + 8);
    }
    #pragma unroll
    for (int j = 0; j < 4; j++) {
      int n = wx * 64 + j * 16 + l15;               // B: lane -> N=l15
      const unsigned* q = Bs32 + n * 16 + half * 4; // K pairs {0..7|8..15}(+16)
      b[j].u[0] = *(const uint4e*)(q);
      b[j].u[1] = *(const uint4e*)(q + 8);
    }
    #pragma unroll
    for (int i = 0; i < 2; i++)
      #pragma unroll
      for (int j = 0; j < 4; j++)
        acc[i][j] = __builtin_amdgcn_wmma_f32_16x16x32_bf16(
            false, a[i].v, false, b[j].v, (short)0, acc[i][j], false, false);

    __syncthreads();
  }

  // ---- store (C layout: elem r -> M=(lane>>4)*8+r, N=lane&15)
  #pragma unroll
  for (int i = 0; i < 2; i++) {
    #pragma unroll
    for (int j = 0; j < 4; j++) {
      int mb = bm + wy * 32 + i * 16 + (lane >> 4) * 8;
      int n  = bn + wx * 64 + j * 16 + (lane & 15);
      if (n < N) {
        float bv = bias ? bias[n] : 0.f;
        #pragma unroll
        for (int r = 0; r < 8; r++) {
          int m = mb + r;
          if (m < M) {
            long long idx = (long long)m * ldc + n;
            float v = acc[i][j][r] * scale + bv;
            if (accum) v += Cf[idx];
            if (do_relu && v < 0.f) v = 0.f;
            if (Cf) Cf[idx] = v; else Cb[idx] = f2bf(v);
          }
        }
      }
    }
  }
}

// ---------------------------------------------------------------------------
// Elementwise / small kernels
// ---------------------------------------------------------------------------
__global__ void k_f32_to_bf16(const float* __restrict__ in, u16* __restrict__ out, long long n) {
  long long i = (long long)blockIdx.x * 256 + threadIdx.x;
  if (i < n) out[i] = f2bf(in[i]);
}

__global__ void k_fill(float* __restrict__ p, float v, long long n) {
  long long i = (long long)blockIdx.x * 256 + threadIdx.x;
  if (i < n) p[i] = v;
}

// hg[b][s][t*4+d] = relu(src[t][b][s] * R_u[s*4+d])   (bf16 out)
__global__ void k_build_hg(const float* __restrict__ src, const float* __restrict__ Ru,
                           u16* __restrict__ hg) {
  long long i = (long long)blockIdx.x * 256 + threadIdx.x;
  if (i >= (long long)B_ * SE_ * DFEAT) return;
  int f = i & 1023;
  int s = (int)((i >> 10) & 127);
  int b = (int)(i >> 17);
  int t = f >> 2, d = f & 3;
  float v = src[((long long)t * B_ + b) * 256 + s] * Ru[s * 4 + d];
  hg[i] = f2bf(v > 0.f ? v : 0.f);
}

// per-row softmax over 128, fused alpha*W -> bf16.  One wave per row.
__global__ __launch_bounds__(128)
void k_prop_softmax(float* __restrict__ sc, const float* __restrict__ W, u16* __restrict__ aw) {
  int row  = blockIdx.x * 4 + (threadIdx.x >> 5);
  int lane = threadIdx.x & 31;
  float* p = sc + (long long)row * 128;
  float v[4], mx = -1e30f;
  #pragma unroll
  for (int j = 0; j < 4; j++) { v[j] = p[lane + 32 * j]; mx = fmaxf(mx, v[j]); }
  mx = wmax(mx);
  float s = 0.f;
  #pragma unroll
  for (int j = 0; j < 4; j++) { v[j] = __expf(v[j] - mx); s += v[j]; }
  s = wsum(s);
  float inv = 1.f / s;
  const float* w = W + (long long)row * 128;
  u16* q = aw + (long long)row * 128;
  #pragma unroll
  for (int j = 0; j < 4; j++) {
    int c = lane + 32 * j;
    float a = v[j] * inv;
    p[c] = a;                       // alpha (f32) -> becomes next layer's W
    q[c] = f2bf(a * w[c]);          // alpha * W_current (bf16 for AV gemm)
  }
}

// xx[t][b][c] : c<512 -> hg_final(f32); c>=512 -> positional encoding
__global__ void k_assemble_xx(const float* __restrict__ cf, const float* __restrict__ times,
                              float* __restrict__ xx) {
  long long i = (long long)blockIdx.x * 256 + threadIdx.x;
  if (i >= (long long)T_ * B_ * DMODEL) return;
  int c = (int)(i % DMODEL);
  long long r = i / DMODEL;
  int b = (int)(r % B_);
  int t = (int)(r / B_);
  float v;
  if (c < 512) {
    v = cf[(long long)b * (SE_ * DFEAT) + (c >> 2) * DFEAT + t * 4 + (c & 3)];
  } else {
    int j = c - 512;
    float e = (float)(j < 8 ? j : j - 8) * (1.f / 7.f);
    float ts = __expf(5.5451774f * e);          // 256^e
    float tt = times[(long long)t * B_ + b];
    v = (j < 8) ? sinf(tt / ts) : cosf(tt / ts);
  }
  xx[i] = v;
}

// qh/kh/vh[b][h][t][d] = qkv[t*B+b][h*132+d (+528,+1056)]
__global__ void k_reshape_heads(const u16* __restrict__ qkv, u16* __restrict__ qh,
                                u16* __restrict__ kh, u16* __restrict__ vh) {
  long long i = (long long)blockIdx.x * 256 + threadIdx.x;
  if (i >= (long long)B_ * NH * T_ * HD) return;
  int d = (int)(i % HD);
  long long r = i / HD;
  int t = (int)(r % T_);
  int z = (int)(r / T_);
  int b = z >> 2, h = z & 3;
  long long src = ((long long)t * B_ + b) * (3 * DMODEL) + h * HD + d;
  qh[i] = qkv[src];
  kh[i] = qkv[src + DMODEL];
  vh[i] = qkv[src + 2 * DMODEL];
}

// masked softmax over T=256, bf16 out. One wave per row; row = (b*4+h)*256+q
__global__ __launch_bounds__(128)
void k_att_softmax(const float* __restrict__ att, u16* __restrict__ out,
                   const int* __restrict__ lengths) {
  int row  = blockIdx.x * 4 + (threadIdx.x >> 5);
  int lane = threadIdx.x & 31;
  int b = row >> 10;                      // 4 heads * 256 queries
  int len = lengths[b];
  const float* p = att + (long long)row * T_;
  float v[8], mx = -1e30f;
  #pragma unroll
  for (int j = 0; j < 8; j++) {
    int k = lane + 32 * j;
    v[j] = (k < len) ? p[k] : -1e9f;
    mx = fmaxf(mx, v[j]);
  }
  mx = wmax(mx);
  float s = 0.f;
  #pragma unroll
  for (int j = 0; j < 8; j++) { v[j] = __expf(v[j] - mx); s += v[j]; }
  s = wsum(s);
  float inv = 1.f / s;
  u16* q = out + (long long)row * T_;
  #pragma unroll
  for (int j = 0; j < 8; j++) q[lane + 32 * j] = f2bf(v[j] * inv);
}

// xx = LayerNorm(xx + o) * g + b   (row length 528, one wave per row)
__global__ __launch_bounds__(128)
void k_add_ln(float* __restrict__ xx, const float* __restrict__ o,
              const float* __restrict__ g, const float* __restrict__ bta) {
  int row  = blockIdx.x * 4 + (threadIdx.x >> 5);
  int lane = threadIdx.x & 31;
  long long base = (long long)row * DMODEL;
  float xs[17];
  float s = 0.f;
  #pragma unroll
  for (int j = 0; j < 17; j++) {
    int c = lane + 32 * j;
    float x = 0.f;
    if (c < DMODEL) x = xx[base + c] + o[base + c];
    xs[j] = x; s += x;
  }
  s = wsum(s);
  float m = s * (1.f / DMODEL);
  float vq = 0.f;
  #pragma unroll
  for (int j = 0; j < 17; j++) {
    int c = lane + 32 * j;
    if (c < DMODEL) { float d = xs[j] - m; vq += d * d; }
  }
  vq = wsum(vq);
  float rs = rsqrtf(vq * (1.f / DMODEL) + 1e-5f);
  #pragma unroll
  for (int j = 0; j < 17; j++) {
    int c = lane + 32 * j;
    if (c < DMODEL) xx[base + c] = (xs[j] - m) * rs * g[c] + bta[c];
  }
}

// pooled[b][528+j] = static @ static_W^T + static_b
__global__ void k_static_emb(const float* __restrict__ st, const float* __restrict__ W,
                             const float* __restrict__ bias, float* __restrict__ pooled) {
  int i = blockIdx.x * 256 + threadIdx.x;
  if (i >= B_ * 512) return;
  int b = i / 512, j = i % 512;
  float s = bias[j];
  for (int k = 0; k < 64; k++) s += st[b * 64 + k] * W[j * 64 + k];
  pooled[(long long)b * DFIN + DMODEL + j] = s;
}

// pooled[b][c<528] = sum_{t<len} xx[t][b][c] / (len+1)
__global__ void k_pool(const float* __restrict__ xx, const int* __restrict__ lengths,
                       float* __restrict__ pooled) {
  int i = blockIdx.x * 256 + threadIdx.x;
  if (i >= B_ * DMODEL) return;
  int b = i / DMODEL, c = i % DMODEL;
  int len = lengths[b];
  float s = 0.f;
  for (int t = 0; t < len; t++) s += xx[((long long)t * B_ + b) * DMODEL + c];
  pooled[(long long)b * DFIN + c] = s / (float)(len + 1);
}

// out[b][j] = hid @ mlp_W2^T + mlp_b2   (N_CLASSES = 2)
__global__ void k_final(const u16* __restrict__ hid, const float* __restrict__ W2,
                        const float* __restrict__ b2, float* __restrict__ out) {
  int i = blockIdx.x * 256 + threadIdx.x;
  if (i >= B_ * 2) return;
  int b = i >> 1, j = i & 1;
  float s = b2[j];
  for (int c = 0; c < DFIN; c++) s += bf2f(hid[(long long)b * DFIN + c]) * W2[j * DFIN + c];
  out[i] = s;
}

// ---------------------------------------------------------------------------
// Host side
// ---------------------------------------------------------------------------
static inline void* wsp(void* ws, size_t off) { return (void*)((char*)ws + off); }
static inline int cdiv(long long a, long long b) { return (int)((a + b - 1) / b); }

static void gemm(hipStream_t st, const void* A, int lda, long long sA,
                 const void* B, int ldb, long long sB, int b_kxn,
                 const float* bias, float* Cf, void* Cb, int ldc, long long sC,
                 int M, int N, int K, float scale, int accum, int relu, int batch) {
  dim3 g(cdiv(N, TN), cdiv(M, TM), batch);
  k_gemm<<<g, 128, 0, st>>>((const u16*)A, lda, sA, (const u16*)B, ldb, sB, b_kxn,
                            bias, Cf, (u16*)Cb, ldc, sC, M, N, K, scale, accum, relu);
}
static void conv(hipStream_t st, const float* in, void* out, long long n) {
  k_f32_to_bf16<<<cdiv(n, 256), 256, 0, st>>>(in, (u16*)out, n);
}

extern "C" void kernel_launch(void* const* d_in, const int* in_sizes, int n_in,
                              void* d_out, int out_size, void* d_ws, size_t ws_size,
                              hipStream_t stream) {
  const float* src      = (const float*)d_in[0];
  const float* times    = (const float*)d_in[1];
  const int*   lengths  = (const int*)  d_in[2];
  const float* stat     = (const float*)d_in[3];
  const float* R_u      = (const float*)d_in[4];
  const float* static_W = (const float*)d_in[5];
  const float* static_b = (const float*)d_in[6];
  const float* p_Wq     = (const float*)d_in[7];
  const float* p_Wk     = (const float*)d_in[8];
  const float* p_Wv     = (const float*)d_in[9];
  const float* p_bq     = (const float*)d_in[10];
  const float* p_bk     = (const float*)d_in[11];
  const float* p_bv     = (const float*)d_in[12];
  const float* p_Wskip  = (const float*)d_in[13];
  const float* p_bskip  = (const float*)d_in[14];
  const float* t_Wqkv   = (const float*)d_in[15];
  const float* t_bqkv   = (const float*)d_in[16];
  const float* t_Wo     = (const float*)d_in[17];
  const float* t_bo     = (const float*)d_in[18];
  const float* t_W1     = (const float*)d_in[19];
  const float* t_b1     = (const float*)d_in[20];
  const float* t_W2     = (const float*)d_in[21];
  const float* t_b2     = (const float*)d_in[22];
  const float* t_ln1g   = (const float*)d_in[23];
  const float* t_ln1b   = (const float*)d_in[24];
  const float* t_ln2g   = (const float*)d_in[25];
  const float* t_ln2b   = (const float*)d_in[26];
  const float* mlp_W1   = (const float*)d_in[27];
  const float* mlp_b1   = (const float*)d_in[28];
  const float* mlp_W2   = (const float*)d_in[29];
  const float* mlp_b2   = (const float*)d_in[30];

  const size_t MBy = 1024ull * 1024ull;
  // persistent
  const size_t O_XX = 0;                     // f32  T*B*528
  const size_t AR   = 70 * MBy;
  // ---- propagation phase arena
  const size_t O_HGA = AR;                   // bf16 B*SE*1024
  const size_t O_HGB = AR + 32 * MBy;
  const size_t O_Q   = AR + 64 * MBy;
  const size_t O_K   = AR + 96 * MBy;
  const size_t O_V   = AR + 128 * MBy;
  const size_t O_SC0 = AR + 160 * MBy;       // f32  B*128*128
  const size_t O_SC1 = AR + 168 * MBy;
  const size_t O_AW  = AR + 176 * MBy;       // bf16 B*128*128
  const size_t O_PW  = AR + 180 * MBy;       // bf16 8 x 1024*1024
  const size_t O_CF  = AR + 196 * MBy;       // f32  B*SE*1024
  // ---- transformer phase arena (reuses the above after the hand-off)
  const size_t O_TW   = AR;                  // bf16 transformer weights
  const size_t O_XXB  = AR + 6 * MBy;        // bf16 T*B*528
  const size_t O_QKV  = AR + 41 * MBy;       // bf16 T*B*1584
  const size_t O_ATTF = AR + 41 * MBy;       // f32  512*256*256 (aliases QKV, dead by then)
  const size_t O_QH   = AR + 176 * MBy;      // bf16 512*256*132
  const size_t O_KH   = AR + 211 * MBy;
  const size_t O_VH   = AR + 246 * MBy;
  const size_t O_ATTB = AR + 281 * MBy;      // bf16 512*256*256
  const size_t O_OF   = AR + 349 * MBy;      // f32  T*B*528
  const size_t O_OB   = AR + 419 * MBy;      // bf16 T*B*528
  const size_t O_FF1  = AR + 454 * MBy;      // bf16 T*B*128
  const size_t O_PLD  = AR + 463 * MBy;      // f32  128*1040
  const size_t O_PLB  = AR + 464 * MBy;      // bf16 128*1040
  const size_t O_HID  = AR + 465 * MBy;      // bf16 128*1040
  const size_t O_MW1  = AR + 466 * MBy;      // bf16 1040*1040

  // ============================ propagation ===============================
  // weights -> bf16: [l][Wq,Wk,Wv,Wskip]
  for (int l = 0; l < 2; l++) {
    const long long WSZ = 1024LL * 1024LL;
    conv(stream, p_Wq    + l * WSZ, wsp(d_ws, O_PW + (size_t)(l * 4 + 0) * WSZ * 2), WSZ);
    conv(stream, p_Wk    + l * WSZ, wsp(d_ws, O_PW + (size_t)(l * 4 + 1) * WSZ * 2), WSZ);
    conv(stream, p_Wv    + l * WSZ, wsp(d_ws, O_PW + (size_t)(l * 4 + 2) * WSZ * 2), WSZ);
    conv(stream, p_Wskip + l * WSZ, wsp(d_ws, O_PW + (size_t)(l * 4 + 3) * WSZ * 2), WSZ);
  }
  k_build_hg<<<cdiv((long long)B_ * SE_ * DFEAT, 256), 256, 0, stream>>>(src, R_u, (u16*)wsp(d_ws, O_HGA));
  k_fill<<<cdiv((long long)B_ * 128 * 128, 256), 256, 0, stream>>>((float*)wsp(d_ws, O_SC0), 1.0f, (long long)B_ * 128 * 128);

  float* Wcur = (float*)wsp(d_ws, O_SC0);
  float* Ascr = (float*)wsp(d_ws, O_SC1);
  const float inv_sqrt_dfeat = 0.03125f;       // 1/sqrt(1024)

  for (int l = 0; l < 2; l++) {
    const void* hg_in = (l == 0) ? wsp(d_ws, O_HGA) : wsp(d_ws, O_HGB);
    const long long WSZ = 1024LL * 1024LL;
    const void* Wq = wsp(d_ws, O_PW + (size_t)(l * 4 + 0) * WSZ * 2);
    const void* Wk = wsp(d_ws, O_PW + (size_t)(l * 4 + 1) * WSZ * 2);
    const void* Wv = wsp(d_ws, O_PW + (size_t)(l * 4 + 2) * WSZ * 2);
    const void* Ws = wsp(d_ws, O_PW + (size_t)(l * 4 + 3) * WSZ * 2);

    // Q/K/V projections: M=16384 N=1024 K=1024, bf16 out
    gemm(stream, hg_in, 1024, 0, Wq, 1024, 0, 0, p_bq + l * 1024, nullptr, wsp(d_ws, O_Q), 1024, 0, 16384, 1024, 1024, 1.f, 0, 0, 1);
    gemm(stream, hg_in, 1024, 0, Wk, 1024, 0, 0, p_bk + l * 1024, nullptr, wsp(d_ws, O_K), 1024, 0, 16384, 1024, 1024, 1.f, 0, 0, 1);
    gemm(stream, hg_in, 1024, 0, Wv, 1024, 0, 0, p_bv + l * 1024, nullptr, wsp(d_ws, O_V), 1024, 0, 16384, 1024, 1024, 1.f, 0, 0, 1);

    // scores: batched (128) 128x128x1024 -> f32
    gemm(stream, wsp(d_ws, O_Q), 1024, 128LL * 1024, wsp(d_ws, O_K), 1024, 128LL * 1024, 0,
         nullptr, Ascr, nullptr, 128, 128LL * 128, 128, 128, 1024, inv_sqrt_dfeat, 0, 0, B_);

    // softmax + alpha*W (bf16)
    k_prop_softmax<<<(B_ * 128) / 4, 128, 0, stream>>>(Ascr, Wcur, (u16*)wsp(d_ws, O_AW));

    // skip path: CF = hg @ Wskip^T + bskip  (f32)
    gemm(stream, hg_in, 1024, 0, Ws, 1024, 0, 0, p_bskip + l * 1024, (float*)wsp(d_ws, O_CF), nullptr, 1024, 0, 16384, 1024, 1024, 1.f, 0, 0, 1);
    // CF += (alpha*W) @ V   batched, B as KxN
    gemm(stream, wsp(d_ws, O_AW), 128, 128LL * 128, wsp(d_ws, O_V), 1024, 128LL * 1024, 1,
         nullptr, (float*)wsp(d_ws, O_CF), nullptr, 1024, 128LL * 1024, 128, 1024, 128, 1.f, 1, 0, B_);

    if (l == 0)
      conv(stream, (const float*)wsp(d_ws, O_CF), wsp(d_ws, O_HGB), (long long)B_ * SE_ * DFEAT);
    { float* t = Wcur; Wcur = Ascr; Ascr = t; }   // W_next = alpha
  }

  // hand-off: xx = [hg reshaped | positional encoding]
  k_assemble_xx<<<cdiv((long long)T_ * B_ * DMODEL, 256), 256, 0, stream>>>(
      (const float*)wsp(d_ws, O_CF), times, (float*)wsp(d_ws, O_XX));

  // ============================ transformer ===============================
  size_t tw = O_TW;
  size_t TWqkv[2], TWo[2], TW1[2], TW2[2];
  for (int l = 0; l < 2; l++) { TWqkv[l] = tw; tw += (size_t)1584 * 528 * 2; }
  for (int l = 0; l < 2; l++) { TWo[l]   = tw; tw += (size_t)528 * 528 * 2; }
  for (int l = 0; l < 2; l++) { TW1[l]   = tw; tw += (size_t)128 * 528 * 2; }
  for (int l = 0; l < 2; l++) { TW2[l]   = tw; tw += (size_t)528 * 128 * 2; }
  for (int l = 0; l < 2; l++) {
    conv(stream, t_Wqkv + (long long)l * 1584 * 528, wsp(d_ws, TWqkv[l]), 1584LL * 528);
    conv(stream, t_Wo   + (long long)l * 528 * 528,  wsp(d_ws, TWo[l]),   528LL * 528);
    conv(stream, t_W1   + (long long)l * 128 * 528,  wsp(d_ws, TW1[l]),   128LL * 528);
    conv(stream, t_W2   + (long long)l * 528 * 128,  wsp(d_ws, TW2[l]),   528LL * 128);
  }

  const int MTOK = T_ * B_;                        // 32768 tokens
  const float inv_sqrt_hd = 0.08703883f;           // 1/sqrt(132)

  for (int l = 0; l < 2; l++) {
    conv(stream, (const float*)wsp(d_ws, O_XX), wsp(d_ws, O_XXB), (long long)MTOK * DMODEL);

    // QKV: 32768 x 1584 x 528 -> bf16
    gemm(stream, wsp(d_ws, O_XXB), DMODEL, 0, wsp(d_ws, TWqkv[l]), DMODEL, 0, 0,
         t_bqkv + l * 1584, nullptr, wsp(d_ws, O_QKV), 1584, 0, MTOK, 1584, DMODEL, 1.f, 0, 0, 1);

    k_reshape_heads<<<cdiv((long long)B_ * NH * T_ * HD, 256), 256, 0, stream>>>(
        (const u16*)wsp(d_ws, O_QKV), (u16*)wsp(d_ws, O_QH), (u16*)wsp(d_ws, O_KH), (u16*)wsp(d_ws, O_VH));

    // scores: 512 x (256 x 256 x 132) -> f32 (overwrites dead QKV region)
    gemm(stream, wsp(d_ws, O_QH), HD, (long long)T_ * HD, wsp(d_ws, O_KH), HD, (long long)T_ * HD, 0,
         nullptr, (float*)wsp(d_ws, O_ATTF), nullptr, T_, (long long)T_ * T_, T_, T_, HD, inv_sqrt_hd, 0, 0, B_ * NH);

    // masked softmax -> bf16
    k_att_softmax<<<(B_ * NH * T_) / 4, 128, 0, stream>>>(
        (const float*)wsp(d_ws, O_ATTF), (u16*)wsp(d_ws, O_ATTB), lengths);

    // AV: writes o(T,B,528) f32 directly (528 == 4*132 so batch stride z*132)
    gemm(stream, wsp(d_ws, O_ATTB), T_, (long long)T_ * T_, wsp(d_ws, O_VH), HD, (long long)T_ * HD, 1,
         nullptr, (float*)wsp(d_ws, O_OF), nullptr, B_ * DMODEL, (long long)HD, T_, HD, T_, 1.f, 0, 0, B_ * NH);

    conv(stream, (const float*)wsp(d_ws, O_OF), wsp(d_ws, O_OB), (long long)MTOK * DMODEL);

    // Wo projection -> f32 into OF
    gemm(stream, wsp(d_ws, O_OB), DMODEL, 0, wsp(d_ws, TWo[l]), DMODEL, 0, 0,
         t_bo + l * DMODEL, (float*)wsp(d_ws, O_OF), nullptr, DMODEL, 0, MTOK, DMODEL, DMODEL, 1.f, 0, 0, 1);

    k_add_ln<<<MTOK / 4, 128, 0, stream>>>((float*)wsp(d_ws, O_XX), (const float*)wsp(d_ws, O_OF),
                                           t_ln1g + l * DMODEL, t_ln1b + l * DMODEL);

    // FFN
    conv(stream, (const float*)wsp(d_ws, O_XX), wsp(d_ws, O_XXB), (long long)MTOK * DMODEL);
    gemm(stream, wsp(d_ws, O_XXB), DMODEL, 0, wsp(d_ws, TW1[l]), DMODEL, 0, 0,
         t_b1 + l * DFF, nullptr, wsp(d_ws, O_FF1), DFF, 0, MTOK, DFF, DMODEL, 1.f, 0, 1, 1);
    gemm(stream, wsp(d_ws, O_FF1), DFF, 0, wsp(d_ws, TW2[l]), DFF, 0, 0,
         t_b2 + l * DMODEL, (float*)wsp(d_ws, O_OF), nullptr, DMODEL, 0, MTOK, DMODEL, DFF, 1.f, 0, 0, 1);

    k_add_ln<<<MTOK / 4, 128, 0, stream>>>((float*)wsp(d_ws, O_XX), (const float*)wsp(d_ws, O_OF),
                                           t_ln2g + l * DMODEL, t_ln2b + l * DMODEL);
  }

  // ============================ head =====================================
  k_static_emb<<<cdiv(B_ * 512, 256), 256, 0, stream>>>(stat, static_W, static_b, (float*)wsp(d_ws, O_PLD));
  k_pool<<<cdiv(B_ * DMODEL, 256), 256, 0, stream>>>((const float*)wsp(d_ws, O_XX), lengths, (float*)wsp(d_ws, O_PLD));

  conv(stream, (const float*)wsp(d_ws, O_PLD), wsp(d_ws, O_PLB), (long long)B_ * DFIN);
  conv(stream, mlp_W1, wsp(d_ws, O_MW1), (long long)DFIN * DFIN);

  // hid = relu(pooled @ W1^T + b1): 128 x 1040 x 1040 -> bf16
  gemm(stream, wsp(d_ws, O_PLB), DFIN, 0, wsp(d_ws, O_MW1), DFIN, 0, 0,
       mlp_b1, nullptr, wsp(d_ws, O_HID), DFIN, 0, B_, DFIN, DFIN, 1.f, 0, 1, 1);

  k_final<<<1, 256, 0, stream>>>((const u16*)wsp(d_ws, O_HID), mlp_W2, mlp_b2, (float*)d_out);
  (void)in_sizes; (void)n_in; (void)out_size; (void)ws_size;
}
